// PointTransformerDecoder_51737176047788
// MI455X (gfx1250) — matrix operand order repaired and checked by reading, not compiled, not run on hardware
//
#include <hip/hip_runtime.h>
#include <hip/hip_bf16.h>

// ---------------------------------------------------------------------------
// MI455X (gfx1250) Point-Transformer decoder.
// Dense layers -> LDS-staged bf16 WMMA GEMM (v_wmma_f32_16x16x32_bf16, wave32):
//   block(256thr/8waves) computes 128x64 C tile; W k-chunk staged in LDS as
//   packed bf16 pairs ([n][kpair] layout). All 8 ds_load_b128 B-fragment loads
//   issued before the 4 WMMAs (separate regs) so dscnt waits are incremental.
// ---------------------------------------------------------------------------

typedef __attribute__((ext_vector_type(16))) __bf16 v16bf;
typedef __attribute__((ext_vector_type(8)))  float  v8f;
typedef __attribute__((ext_vector_type(8)))  unsigned v8u;

__device__ __forceinline__ unsigned pack_bf16(float x, float y) {
#if __has_builtin(__builtin_amdgcn_cvt_pk_bf16_f32)
  auto p = __builtin_amdgcn_cvt_pk_bf16_f32(x, y);
  return __builtin_bit_cast(unsigned, p);
#else
  unsigned ux = __builtin_bit_cast(unsigned, x) + 0x8000u;  // round-to-nearest
  unsigned uy = __builtin_bit_cast(unsigned, y) + 0x8000u;
  // result = { bf16(y), bf16(x) } : bytes [ux2 ux3 uy2 uy3]
  return __builtin_amdgcn_perm(uy, ux, 0x07060302u);
#endif
}

__device__ __forceinline__ v16bf mk_b(uint4 lo, uint4 hi) {
  v8u bu;
  bu[0] = lo.x; bu[1] = lo.y; bu[2] = lo.z; bu[3] = lo.w;
  bu[4] = hi.x; bu[5] = hi.y; bu[6] = hi.z; bu[7] = hi.w;
  return __builtin_bit_cast(v16bf, bu);
}

#define WMMA_BF16(A_, B_, C_) \
  __builtin_amdgcn_wmma_f32_16x16x32_bf16(false, A_, false, B_, (short)0, C_, false, false)

// ---------------------------------------------------------------------------
// C[M,N] = act(A[M,K] @ W[K,N] + bias (+ residual))
// M,N multiples of 16, K multiple of 32 (true for every layer here).
// Block = 8 waves -> 128(M) x 64(N) tile; wave w -> rows 16w..16w+15.
// ---------------------------------------------------------------------------
__global__ void k_gemm(const float* __restrict__ A, const float* __restrict__ W,
                       const float* __restrict__ bias, const float* __restrict__ resid,
                       float* __restrict__ C, int M, int N, int Kd, int relu)
{
  __shared__ __align__(16) unsigned lb[2][64 * 16];  // [buf][n*16 + kpair]
  int tid  = threadIdx.x;
  int wave = tid >> 5, lane = tid & 31;
  int half = lane >> 4, l16 = lane & 15;

  int bM  = (M + 127) >> 7;
  int bm  = blockIdx.x % bM;
  int bn  = blockIdx.x / bM;
  int cn0 = bn << 6;

  int row0  = (bm << 7) + wave * 16;
  int rowOK = (row0 < M);                 // wave-uniform
  if (!rowOK) row0 = M - 16;              // clamp; stores guarded off

  v8f acc0 = {}, acc1 = {}, acc2 = {}, acc3 = {};
  const float* arow = A + (size_t)(row0 + l16) * Kd;

  int nch = Kd >> 5;
  for (int c = 0; c < nch; ++c) {
    int k0 = c << 5;
    __builtin_prefetch(arow + k0 + 32, 0, 1);
    unsigned* Lb = lb[c & 1];
    // ---- cooperative fill: 1024 packed entries, 4 per thread, coalesced in n
#pragma unroll
    for (int e = 0; e < 4; ++e) {
      int idx = tid + (e << 8);           // 0..1023
      int n   = idx & 63;
      int kp  = idx >> 6;                 // 0..15
      int col = cn0 + n; if (col >= N) col = N - 1;
      float f0 = W[(size_t)(k0 + 2 * kp)     * N + col];
      float f1 = W[(size_t)(k0 + 2 * kp + 1) * N + col];
      Lb[n * 16 + kp] = pack_bf16(f0, f1);
    }
    // ---- A fragment (16x32 bf16): two 8-float contiguous runs per lane
    const float* ar = arow + k0 + half * 8;
    float4 a0 = *(const float4*)(ar);
    float4 a1 = *(const float4*)(ar + 4);
    float4 a2 = *(const float4*)(ar + 16);
    float4 a3 = *(const float4*)(ar + 20);
    v8u au;
    au[0] = pack_bf16(a0.x, a0.y); au[1] = pack_bf16(a0.z, a0.w);
    au[2] = pack_bf16(a1.x, a1.y); au[3] = pack_bf16(a1.z, a1.w);
    au[4] = pack_bf16(a2.x, a2.y); au[5] = pack_bf16(a2.z, a2.w);
    au[6] = pack_bf16(a3.x, a3.y); au[7] = pack_bf16(a3.z, a3.w);
    v16bf a = __builtin_bit_cast(v16bf, au);
    __syncthreads();                      // fill visible; prev consume retired
    // ---- issue all 8 ds_load_b128 first (separate regs), then 4 WMMAs
    const unsigned* p0 = Lb + (l16 << 4) + half * 4;   // column l16, subtile 0
    uint4 b0l = *(const uint4*)(p0);
    uint4 b0h = *(const uint4*)(p0 + 8);
    uint4 b1l = *(const uint4*)(p0 + 256);
    uint4 b1h = *(const uint4*)(p0 + 264);
    uint4 b2l = *(const uint4*)(p0 + 512);
    uint4 b2h = *(const uint4*)(p0 + 520);
    uint4 b3l = *(const uint4*)(p0 + 768);
    uint4 b3h = *(const uint4*)(p0 + 776);
    acc0 = WMMA_BF16(a, mk_b(b0l, b0h), acc0);
    acc1 = WMMA_BF16(a, mk_b(b1l, b1h), acc1);
    acc2 = WMMA_BF16(a, mk_b(b2l, b2h), acc2);
    acc3 = WMMA_BF16(a, mk_b(b3l, b3h), acc3);
  }

  // ---- store (wave-uniform guards only): dword r -> row row0+half*8+r
  if (rowOK) {
#pragma unroll
    for (int j = 0; j < 4; ++j) {
      int ct = cn0 + j * 16;
      if (ct < N) {
        v8f accv = (j == 0) ? acc0 : (j == 1) ? acc1 : (j == 2) ? acc2 : acc3;
        int col = ct + l16;
        float bv = bias ? bias[col] : 0.f;
#pragma unroll
        for (int r = 0; r < 8; ++r) {
          int row = row0 + half * 8 + r;
          float v = accv[r] + bv;
          if (resid) v += resid[(size_t)row * N + col];
          if (relu)  v = fmaxf(v, 0.f);
          C[(size_t)row * N + col] = v;
        }
      }
    }
  }
}

// out[r,c] = relu(in[r,0:3] @ W[3,N] + b)   (K=3 too small for WMMA)
__global__ void k_lin3_relu(const float* __restrict__ in, const float* __restrict__ W,
                            const float* __restrict__ b, float* __restrict__ out,
                            int Mrows, int N)
{
  int i = blockIdx.x * blockDim.x + threadIdx.x;
  if (i >= Mrows * N) return;
  int r = i / N, c = i - r * N;
  const float* rr = in + (size_t)r * 3;
  float v = b[c] + rr[0] * W[c] + rr[1] * W[N + c] + rr[2] * W[2 * N + c];
  out[i] = fmaxf(v, 0.f);
}

// kNN-16 (self included), insertion top-k per query point
__global__ void k_knn16(const float* __restrict__ xyz, int B, int N, int* __restrict__ idx)
{
  int t = blockIdx.x * blockDim.x + threadIdx.x;
  if (t >= B * N) return;
  int b = t / N, n = t - b * N;
  const float* base = xyz + (size_t)b * N * 3;
  float qx = base[n * 3], qy = base[n * 3 + 1], qz = base[n * 3 + 2];
  float dist[16]; int di[16];
#pragma unroll
  for (int j = 0; j < 16; ++j) { dist[j] = 3.0e38f; di[j] = 0; }
  for (int m = 0; m < N; ++m) {
    float dx = base[m * 3] - qx, dy = base[m * 3 + 1] - qy, dz = base[m * 3 + 2] - qz;
    float d = dx * dx + dy * dy + dz * dz;
    if (d < dist[15]) {
      int p = 15;
      while (p > 0 && dist[p - 1] > d) { dist[p] = dist[p - 1]; di[p] = di[p - 1]; --p; }
      dist[p] = d; di[p] = m;
    }
  }
#pragma unroll
  for (int j = 0; j < 16; ++j) idx[(size_t)t * 16 + j] = di[j];
}

// 3-NN fine->coarse + normalized inverse-distance weights
__global__ void k_knn3(const float* __restrict__ xf, const float* __restrict__ xc,
                       int B, int Nf, int Nc, int* __restrict__ id3, float* __restrict__ w3)
{
  int t = blockIdx.x * blockDim.x + threadIdx.x;
  if (t >= B * Nf) return;
  int b = t / Nf, n = t - b * Nf;
  const float* q = xf + ((size_t)b * Nf + n) * 3;
  const float* base = xc + (size_t)b * Nc * 3;
  float d0 = 3e38f, d1 = 3e38f, d2 = 3e38f; int i0 = 0, i1 = 0, i2 = 0;
  for (int m = 0; m < Nc; ++m) {
    float dx = base[m * 3] - q[0], dy = base[m * 3 + 1] - q[1], dz = base[m * 3 + 2] - q[2];
    float d = dx * dx + dy * dy + dz * dz;
    if (d < d0)      { d2 = d1; i2 = i1; d1 = d0; i1 = i0; d0 = d; i0 = m; }
    else if (d < d1) { d2 = d1; i2 = i1; d1 = d; i1 = m; }
    else if (d < d2) { d2 = d; i2 = m; }
  }
  float w0 = 1.f / (d0 + 1e-8f), w1 = 1.f / (d1 + 1e-8f), w2 = 1.f / (d2 + 1e-8f);
  float s = w0 + w1 + w2;
  id3[t * 3] = i0; id3[t * 3 + 1] = i1; id3[t * 3 + 2] = i2;
  w3[t * 3] = w0 / s; w3[t * 3 + 1] = w1 / s; w3[t * 3 + 2] = w2 / s;
}

// interp = sum_j w_j * f1[idx_j] + f2
__global__ void k_interp_add(const float* __restrict__ f1, const float* __restrict__ f2,
                             const int* __restrict__ id3, const float* __restrict__ w3,
                             float* __restrict__ out, int B, int Nf, int Nc, int C)
{
  int t = blockIdx.x * blockDim.x + threadIdx.x;
  if (t >= B * Nf * C) return;
  int c = t % C; int r = t / C; int b = r / Nf;
  const int* id = id3 + (size_t)r * 3;
  const float* w = w3 + (size_t)r * 3;
  const float* fb = f1 + (size_t)b * Nc * C;
  float v = w[0] * fb[(size_t)id[0] * C + c]
          + w[1] * fb[(size_t)id[1] * C + c]
          + w[2] * fb[(size_t)id[2] * C + c];
  out[t] = v + f2[t];
}

// training-mode BatchNorm1d over rows, fused ReLU; one block per channel
__global__ void k_bn_relu(float* __restrict__ x, const float* __restrict__ g,
                          const float* __restrict__ bt, int M, int C)
{
  int c = blockIdx.x;
  __shared__ float ss[256], ss2[256];
  float s = 0.f, s2 = 0.f;
  for (int r = threadIdx.x; r < M; r += blockDim.x) {
    float v = x[(size_t)r * C + c]; s += v; s2 += v * v;
  }
  ss[threadIdx.x] = s; ss2[threadIdx.x] = s2; __syncthreads();
  for (int o = 128; o > 0; o >>= 1) {
    if ((int)threadIdx.x < o) { ss[threadIdx.x] += ss[threadIdx.x + o]; ss2[threadIdx.x] += ss2[threadIdx.x + o]; }
    __syncthreads();
  }
  float mean = ss[0] / M;
  float var  = ss2[0] / M - mean * mean;
  float rs   = rsqrtf(var + 1e-5f);
  float gg = g[c], bb = bt[c];
  for (int r = threadIdx.x; r < M; r += blockDim.x) {
    size_t o = (size_t)r * C + c;
    x[o] = fmaxf((x[o] - mean) * rs * gg + bb, 0.f);
  }
}

// rel[b,n,k,:] = xyz[b,n] - xyz[b,idx[b,n,k]]
__global__ void k_rel(const float* __restrict__ xyz, const int* __restrict__ idx,
                      float* __restrict__ rel, int B, int N)
{
  int t = blockIdx.x * blockDim.x + threadIdx.x;
  if (t >= B * N * 16) return;
  int bn = t >> 4; int b = bn / N; int n = bn - b * N;
  const float* base = xyz + (size_t)b * N * 3;
  int m = idx[t];
  rel[(size_t)t * 3]     = base[n * 3]     - base[m * 3];
  rel[(size_t)t * 3 + 1] = base[n * 3 + 1] - base[m * 3 + 1];
  rel[(size_t)t * 3 + 2] = base[n * 3 + 2] - base[m * 3 + 2];
}

// attin[b,n,k,c] = q[b,n,c] - K[b,idx,c] + pos[b,n,k,c]     (dm = 256)
__global__ void k_combine(const float* __restrict__ q, const float* __restrict__ kmat,
                          const float* __restrict__ pos, const int* __restrict__ idx,
                          float* __restrict__ out, int B, int N)
{
  int t = blockIdx.x * blockDim.x + threadIdx.x;
  if (t >= B * N * 16 * 256) return;
  int c = t & 255; int bnk = t >> 8; int bn = bnk >> 4; int b = bn / N;
  int m = idx[bnk];
  out[t] = q[(size_t)bn * 256 + c] - kmat[((size_t)b * N + m) * 256 + c] + pos[t];
}

// softmax over K=16 (per (b,n,c), scale 1/sqrt(256)) fused with weighted sum
__global__ void k_softmax_res(const float* __restrict__ a, const float* __restrict__ vmat,
                              const float* __restrict__ pos, const int* __restrict__ idx,
                              float* __restrict__ res, int B, int N)
{
  int t = blockIdx.x * blockDim.x + threadIdx.x;
  if (t >= B * N * 256) return;
  int c = t & 255; int bn = t >> 8; int b = bn / N;
  const int* id = idx + (size_t)bn * 16;
  float mx = -3e38f;
#pragma unroll
  for (int k = 0; k < 16; ++k)
    mx = fmaxf(mx, a[((size_t)bn * 16 + k) * 256 + c] * 0.0625f);
  float se = 0.f, acc = 0.f;
#pragma unroll
  for (int k = 0; k < 16; ++k) {
    float l = a[((size_t)bn * 16 + k) * 256 + c] * 0.0625f;
    float e = __expf(l - mx);
    float val = vmat[((size_t)b * N + id[k]) * 256 + c] + pos[((size_t)bn * 16 + k) * 256 + c];
    se += e; acc += e * val;
  }
  res[t] = acc / se;
}

// ---------------------------------------------------------------------------
// Host-side orchestration
// ---------------------------------------------------------------------------
static inline int ceildiv(int a, int b) { return (a + b - 1) / b; }

static void gemm(const float* A, const float* W, const float* bias, const float* resid,
                 float* C, int M, int N, int Kd, int relu, hipStream_t s)
{
  int bMn = ceildiv(M, 128), bNn = ceildiv(N, 64);
  k_gemm<<<bMn * bNn, 256, 0, s>>>(A, W, bias, resid, C, M, N, Kd, relu);
}

struct TBp { const float *d1w,*d1b,*d2w,*d2b,*fc1w,*fc1b,*fc2w,*fc2b,
                          *g1w,*g1b,*g2w,*g2b,*wk,*wq,*wv; };
struct TUPp { const float *bn1g,*bn1b,*bn2g,*bn2b,*fc1w,*fc1b,*fc2w,*fc2b; };

struct WSb {
  int*   idx;  int* idx3;  float* w3;
  float *x1, *q, *k, *v, *r, *rel, *pos, *t1, *t2, *f1, *f2, *pA, *pB;
};

static void transformer(const float* xyz, const float* feats, float* outpts,
                        const TBp& p, int B, int N, int d, const WSb& ws, hipStream_t s)
{
  int BN = B * N, BNK = BN * 16;
  k_knn16<<<ceildiv(BN, 256), 256, 0, s>>>(xyz, B, N, ws.idx);
  gemm(feats, p.fc1w, p.fc1b, nullptr, ws.x1, BN, 256, d, 0, s);
  gemm(ws.x1, p.wq, nullptr, nullptr, ws.q, BN, 256, 256, 0, s);
  gemm(ws.x1, p.wk, nullptr, nullptr, ws.k, BN, 256, 256, 0, s);
  gemm(ws.x1, p.wv, nullptr, nullptr, ws.v, BN, 256, 256, 0, s);
  k_rel<<<ceildiv(BNK, 256), 256, 0, s>>>(xyz, ws.idx, ws.rel, B, N);
  k_lin3_relu<<<ceildiv(BNK * 256, 256), 256, 0, s>>>(ws.rel, p.d1w, p.d1b, ws.t1, BNK, 256);
  gemm(ws.t1, p.d2w, p.d2b, nullptr, ws.pos, BNK, 256, 256, 0, s);
  k_combine<<<ceildiv(BNK * 256, 256), 256, 0, s>>>(ws.q, ws.k, ws.pos, ws.idx, ws.t1, B, N);
  gemm(ws.t1, p.g1w, p.g1b, nullptr, ws.t2, BNK, 256, 256, 1, s);
  gemm(ws.t2, p.g2w, p.g2b, nullptr, ws.t1, BNK, 256, 256, 0, s);
  k_softmax_res<<<ceildiv(BN * 256, 256), 256, 0, s>>>(ws.t1, ws.v, ws.pos, ws.idx, ws.r, B, N);
  gemm(ws.r, p.fc2w, p.fc2b, feats, outpts, BN, d, 256, 0, s);
}

static void transition_up(const float* xyz_c, const float* pts_c, int Nc, int dc,
                          const float* xyz_f, const float* feat_f, int Nf, int ch,
                          const TUPp& p, float* out, const WSb& ws, int B, hipStream_t s)
{
  gemm(pts_c, p.fc1w, p.fc1b, nullptr, ws.f1, B * Nc, ch, dc, 0, s);
  k_bn_relu<<<ch, 256, 0, s>>>(ws.f1, p.bn1g, p.bn1b, B * Nc, ch);
  gemm(feat_f, p.fc2w, p.fc2b, nullptr, ws.f2, B * Nf, ch, ch, 0, s);
  k_bn_relu<<<ch, 256, 0, s>>>(ws.f2, p.bn2g, p.bn2b, B * Nf, ch);
  k_knn3<<<ceildiv(B * Nf, 256), 256, 0, s>>>(xyz_f, xyz_c, B, Nf, Nc, ws.idx3, ws.w3);
  k_interp_add<<<ceildiv(B * Nf * ch, 256), 256, 0, s>>>(ws.f1, ws.f2, ws.idx3, ws.w3,
                                                         out, B, Nf, Nc, ch);
}

extern "C" void kernel_launch(void* const* d_in, const int* in_sizes, int n_in,
                              void* d_out, int out_size, void* d_ws, size_t ws_size,
                              hipStream_t stream)
{
  (void)in_sizes; (void)n_in; (void)out_size; (void)ws_size;
  const int B = 2;
  const int Ns[5]   = {4096, 1024, 256, 64, 16};
  const int dims[5] = {32, 64, 128, 256, 512};
  auto F = [&](int i) { return (const float*)d_in[i]; };

  // ---- parameter indexing (jax pytree flatten: sorted dict keys, list order)
  // 10..15: fc[0..2] (w,b);  16..30: tb_top;  31+15*i: tbs[i];  91+8*i: tups[i]
  auto loadTB = [&](int bi) {
    TBp t;
    t.d1w = F(bi + 0);  t.d1b = F(bi + 1);  t.d2w = F(bi + 2);  t.d2b = F(bi + 3);
    t.fc1w = F(bi + 4); t.fc1b = F(bi + 5); t.fc2w = F(bi + 6); t.fc2b = F(bi + 7);
    t.g1w = F(bi + 8);  t.g1b = F(bi + 9);  t.g2w = F(bi + 10); t.g2b = F(bi + 11);
    t.wk = F(bi + 12);  t.wq = F(bi + 13);  t.wv = F(bi + 14);
    return t;
  };
  auto loadTUP = [&](int bi) {
    TUPp t;
    t.bn1g = F(bi + 0); t.bn1b = F(bi + 1); t.bn2g = F(bi + 2); t.bn2b = F(bi + 3);
    t.fc1w = F(bi + 4); t.fc1b = F(bi + 5); t.fc2w = F(bi + 6); t.fc2b = F(bi + 7);
    return t;
  };

  // ---- workspace bump allocator
  const int BNmax = B * 4096, BNKmax = BNmax * 16;
  float* base = (float*)d_ws;
  size_t off = 0;
  auto alloc = [&](size_t n) { float* r = base + off; off += (n + 63) & ~(size_t)63; return r; };
  WSb ws;
  ws.idx  = (int*)alloc(BNKmax);
  ws.idx3 = (int*)alloc((size_t)BNmax * 3);
  ws.w3   = alloc((size_t)BNmax * 3);
  ws.x1   = alloc((size_t)BNmax * 256);
  ws.q    = alloc((size_t)BNmax * 256);
  ws.k    = alloc((size_t)BNmax * 256);
  ws.v    = alloc((size_t)BNmax * 256);
  ws.r    = alloc((size_t)BNmax * 256);
  ws.rel  = alloc((size_t)BNKmax * 3);
  ws.pos  = alloc((size_t)BNKmax * 256);
  ws.t1   = alloc((size_t)BNKmax * 256);
  ws.t2   = alloc((size_t)BNKmax * 256);
  ws.f1   = alloc((size_t)BNmax * 256);
  ws.f2   = alloc((size_t)BNmax * 256);
  ws.pA   = alloc(512 * 1024);
  ws.pB   = alloc(512 * 1024);

  // ---- stage A: 3-layer MLP on coarse points [B*16, 512]
  const float* points = F(0);
  const float* xyz4   = F(1);
  float* m0 = ws.f1;   // small scratch reuse ([32,256] / [32,512])
  float* m1 = ws.f2;
  gemm(points, F(10), F(11), nullptr, m0, B * 16, 256, 512, 1, stream);
  gemm(m0,     F(12), F(13), nullptr, m1, B * 16, 256, 256, 1, stream);
  gemm(m1,     F(14), F(15), nullptr, ws.pA, B * 16, 512, 256, 0, stream);

  // ---- top transformer (N=16, d=512, dm=256)
  TBp tbtop = loadTB(16);
  transformer(xyz4, ws.pA, ws.pB, tbtop, B, 16, 512, ws, stream);

  // ---- 4 decode levels: transition_up + transformer
  const float* curxyz = xyz4;
  float* cur = ws.pB;
  int curN = 16, curd = 512;
  for (int i = 0; i < 4; ++i) {
    int lvl = 3 - i;                      // fine level: 3,2,1,0
    const float* xf = F(2 + 2 * i);
    const float* ff = F(3 + 2 * i);
    int Nf = Ns[lvl], ch = dims[lvl];
    TUPp tup = loadTUP(91 + 8 * i);
    transition_up(curxyz, cur, curN, curd, xf, ff, Nf, ch, tup, ws.pA, ws, B, stream);
    TBp tb = loadTB(31 + 15 * i);
    float* dst = (i == 3) ? (float*)d_out : ws.pB;
    transformer(xf, ws.pA, dst, tb, B, Nf, ch, ws, stream);
    curxyz = xf; cur = dst; curN = Nf; curd = ch;
  }
}
